// GivensSSM_53661321396647
// MI455X (gfx1250) — matrix operand order repaired and checked by reading, not compiled, not run on hardware
//
#include <hip/hip_runtime.h>
#include <hip/hip_bf16.h>

// ---------------------------------------------------------------------------
// GivensSSM fused kernel set for gfx1250 (MI455X).
//   D_MODEL = 2048, N_PAIRS = 1024, BATCH = 16384
// bf16 WMMA (v_wmma_f32_16x16x32_bf16) GEMM with fp32 accumulate, fused
// Givens-rotation epilogue. Weights pre-converted to bf16 in d_ws; x is
// converted fp32->bf16 on the fly (x fp32 + bf16 weights fit in 192MB L2).
// K-tile = 64 (two WMMA K-halves per LDS buffer) to halve barrier count;
// all three B fragments are loaded into distinct registers before the WMMA
// triple so the matrix pipe issues back-to-back.
// ---------------------------------------------------------------------------

#define D_MODEL 2048
#define N_PAIRS 1024
#define BATCH   16384

typedef __attribute__((ext_vector_type(16))) __bf16 v16bf;
typedef __attribute__((ext_vector_type(8)))  __bf16 v8bf;
typedef __attribute__((ext_vector_type(4)))  __bf16 v4bf;
typedef __attribute__((ext_vector_type(8)))  float  v8f;

union V16U { v16bf v; v8bf h[2]; __bf16 e[16]; };

// LDS B-tile: 48 rows (16 angle + 32 injection cols) x 64 K, padded row
// stride of 72 bf16 (144 B). 144B row advance = 36 banks (gcd(36,64)=4):
// 16 lanes hit 16 distinct 4-bank groups -> conflict-free ds_load_b128.
#define LDS_STRIDE 72

// ---------------------------------------------------------------------------
// Kernel 1: convert W_angle [1024x2048] and W_in [2048x2048] fp32 -> bf16.
// ---------------------------------------------------------------------------
__global__ __launch_bounds__(256) void givens_wconv_kernel(
    const float* __restrict__ wang, const float* __restrict__ win,
    __bf16* __restrict__ wangBf, __bf16* __restrict__ winBf) {
  const size_t NA = (size_t)N_PAIRS * D_MODEL;           // 2,097,152
  size_t i4 = ((size_t)blockIdx.x * blockDim.x + threadIdx.x) * 4;
  const float* src;
  __bf16* dst;
  if (i4 < NA) { src = wang + i4; dst = wangBf + i4; }
  else         { src = win + (i4 - NA); dst = winBf + (i4 - NA); }
  float4 f = *(const float4*)src;
  v4bf o;
  o[0] = (__bf16)f.x; o[1] = (__bf16)f.y; o[2] = (__bf16)f.z; o[3] = (__bf16)f.w;
  *(v4bf*)dst = o;                                        // 8-byte store
}

// ---------------------------------------------------------------------------
// Kernel 2: decay[b] = sigmoid(x[b,:] . W_decay + b_decay). Wave per row.
// ---------------------------------------------------------------------------
__global__ __launch_bounds__(256) void givens_decay_kernel(
    const float* __restrict__ x, const float* __restrict__ wd,
    const float* __restrict__ bd, float* __restrict__ decay) {
  const int lane = threadIdx.x & 31;
  const int w    = threadIdx.x >> 5;
  const int row  = blockIdx.x * 8 + w;
  const float* xr = x + (size_t)row * D_MODEL;
  float acc = 0.f;
#pragma unroll
  for (int j = 0; j < 16; ++j) {
    int e = (j * 32 + lane) * 4;                          // coalesced float4
    float4 xv = *(const float4*)(xr + e);
    float4 wv = *(const float4*)(wd + e);
    acc += xv.x * wv.x + xv.y * wv.y + xv.z * wv.z + xv.w * wv.w;
  }
#pragma unroll
  for (int m = 16; m >= 1; m >>= 1) acc += __shfl_xor(acc, m, 32);
  if (lane == 0) decay[row] = 1.f / (1.f + __expf(-(acc + bd[0])));
}

// ---------------------------------------------------------------------------
// Kernel 3: fused GEMM (angles + injection) + Givens rotation epilogue.
// Grid: (N_PAIRS/16, BATCH/128), 256 threads (8 waves).
// Wave w owns M-subtile rows [m0, m0+16) and 3 WMMA accumulators:
//   accs[0] : angles cols  p0..p0+15
//   accs[1] : injection cols 2p0..2p0+15
//   accs[2] : injection cols 2p0+16..2p0+31
// ---------------------------------------------------------------------------
__global__ __launch_bounds__(256) void givens_gemm_kernel(
    const float*  __restrict__ x,      const float* __restrict__ h_prev,
    const __bf16* __restrict__ wangBf, const __bf16* __restrict__ winBf,
    const float*  __restrict__ b_angle,const float* __restrict__ b_in,
    const float*  __restrict__ decay,  float* __restrict__ out) {
  const int lane = threadIdx.x & 31;
  const int w    = threadIdx.x >> 5;
  const int nl   = lane & 15;
  const int hi   = lane >> 4;
  const int p0   = blockIdx.x * 16;
  const int m0   = blockIdx.y * 128 + w * 16;

  __shared__ __bf16 ldsB[2][48 * LDS_STRIDE];

  v8f accs[3];
  accs[0] = (v8f){}; accs[1] = (v8f){}; accs[2] = (v8f){};

  // A fragment source (fp32 x, converted on the fly). Layout for 16-bit A:
  // lanes 0-15 hold K {0..7,16..23}, lanes 16-31 hold K {8..15,24..31}.
  const float* aBase = x + (size_t)(m0 + nl) * D_MODEL + hi * 8;

  // Cooperative B-tile staging: 192 threads x 32B (two 16B chunks) per step.
  const int  t       = threadIdx.x;
  const bool bActive = t < 192;
  const int  brow    = t >> 2;
  const int  bq      = t & 3;
  const __bf16* bSrc =
      (brow < 16) ? wangBf + (size_t)(p0 + brow)          * D_MODEL + bq * 8
                  : winBf  + (size_t)(2 * p0 + brow - 16) * D_MODEL + bq * 8;
  const int ldsOff = brow * LDS_STRIDE + bq * 8;

  v8bf breg0, breg1;
  if (bActive) {
    breg0 = *(const v8bf*)(bSrc);
    breg1 = *(const v8bf*)(bSrc + 32);
    *(v8bf*)(&ldsB[0][ldsOff])      = breg0;
    *(v8bf*)(&ldsB[0][ldsOff + 32]) = breg1;
  }

  const int KT = D_MODEL / 64;                            // 32 K-steps of 64
  for (int kt = 0; kt < KT; ++kt) {
    __syncthreads();
    const int cur = kt & 1;
    // Prefetch next B tile (global -> reg) while computing current.
    if (bActive && (kt + 1) < KT) {
      breg0 = *(const v8bf*)(bSrc + (kt + 1) * 64);
      breg1 = *(const v8bf*)(bSrc + (kt + 1) * 64 + 32);
    }

    const __bf16* bp = &ldsB[cur][0];
#pragma unroll
    for (int kh = 0; kh < 2; ++kh) {
      // A fragment: 16 fp32 -> 16 bf16 in WMMA A layout.
      const float* ap = aBase + kt * 64 + kh * 32;
      float fa[16];
      *(float4*)&fa[0]  = *(const float4*)(ap);
      *(float4*)&fa[4]  = *(const float4*)(ap + 4);
      *(float4*)&fa[8]  = *(const float4*)(ap + 16);
      *(float4*)&fa[12] = *(const float4*)(ap + 20);
      V16U au;
#pragma unroll
      for (int i = 0; i < 16; ++i) au.e[i] = (__bf16)fa[i];

      // Load ALL three B fragments first (distinct regs) so the three
      // WMMAs below issue back-to-back instead of stalling on dscnt each.
      V16U bu[3];
#pragma unroll
      for (int ns = 0; ns < 3; ++ns) {
        // B layout: lanes 0-15 K {0..7,16..23}, lanes 16-31 K {8..15,24..31}, col = lane&15.
        const __bf16* q = bp + (ns * 16 + nl) * LDS_STRIDE + kh * 32 + hi * 8;
        bu[ns].h[0] = *(const v8bf*)q;
        bu[ns].h[1] = *(const v8bf*)(q + 16);
      }
#pragma unroll
      for (int ns = 0; ns < 3; ++ns) {
        accs[ns] = __builtin_amdgcn_wmma_f32_16x16x32_bf16(
            false, au.v, false, bu[ns].v, (short)0, accs[ns], false, false);
      }
    }

    if (bActive && (kt + 1) < KT) {
      *(v8bf*)(&ldsB[cur ^ 1][ldsOff])      = breg0;
      *(v8bf*)(&ldsB[cur ^ 1][ldsOff + 32]) = breg1;
    }
  }

  // ---------------- Epilogue: Givens rotate + decay + injection -----------
  // C/D layout: col N = lane&15; row M = vgpr + 8*(lane>>4).
  const float aB  = b_angle[p0 + nl];
  const float bi0 = b_in[2 * p0 + nl];
  const float bi1 = b_in[2 * p0 + 16 + nl];

#pragma unroll
  for (int j = 0; j < 8; ++j) {
    const int m = m0 + j + 8 * hi;
    const float ang = accs[0][j] + aB;                    // pair p0+nl, row m
    const float cs  = __cosf(ang);
    const float sn  = __sinf(ang);
    const float dm  = decay[m];
    const float* hrow = h_prev + (size_t)m * D_MODEL;
    float*       orow = out    + (size_t)m * D_MODEL;
#pragma unroll
    for (int tt = 0; tt < 2; ++tt) {
      const int c   = 2 * p0 + 16 * tt + nl;              // output column
      const int src = (hi << 4) + 8 * tt + (nl >> 1);     // angle-owner lane
      const float csS = __shfl(cs, src, 32);
      const float snS = __shfl(sn, src, 32);
      const float2 hv = *(const float2*)(hrow + (c & ~1));
      const float inj = ((tt == 0) ? accs[1][j] : accs[2][j]) +
                        ((tt == 0) ? bi0 : bi1);
      const float rot = ((c & 1) == 0) ? (csS * hv.x - snS * hv.y)
                                       : (snS * hv.x + csS * hv.y);
      orow[c] = dm * rot + inj;
    }
  }
}

// ---------------------------------------------------------------------------
// Launcher. Inputs: 0:x 1:h_prev 2:W_angle 3:b_angle 4:W_decay 5:b_decay
//                   6:W_in 7:b_in.  Output: [BATCH, D_MODEL] fp32.
// ---------------------------------------------------------------------------
extern "C" void kernel_launch(void* const* d_in, const int* in_sizes, int n_in,
                              void* d_out, int out_size, void* d_ws, size_t ws_size,
                              hipStream_t stream) {
  const float* x       = (const float*)d_in[0];
  const float* h_prev  = (const float*)d_in[1];
  const float* W_angle = (const float*)d_in[2];
  const float* b_angle = (const float*)d_in[3];
  const float* W_decay = (const float*)d_in[4];
  const float* b_decay = (const float*)d_in[5];
  const float* W_in    = (const float*)d_in[6];
  const float* b_in    = (const float*)d_in[7];
  float* out = (float*)d_out;

  // Workspace layout (~12.1 MB).
  char* ws = (char*)d_ws;
  __bf16* wangBf = (__bf16*)(ws);                                   //  4 MiB
  __bf16* winBf  = (__bf16*)(ws + (size_t)N_PAIRS * D_MODEL * 2);   //  8 MiB
  float*  decay  = (float*)(ws + (size_t)(N_PAIRS + D_MODEL) * D_MODEL * 2);

  // 1) Weight fp32 -> bf16 (6,291,456 elems, 4 per thread).
  {
    int total4 = (N_PAIRS * D_MODEL + D_MODEL * D_MODEL) / 4;
    givens_wconv_kernel<<<total4 / 256, 256, 0, stream>>>(
        W_angle, W_in, wangBf, winBf);
  }
  // 2) Decay GEMV (wave per row).
  givens_decay_kernel<<<BATCH / 8, 256, 0, stream>>>(x, W_decay, b_decay, decay);
  // 3) Fused GEMM + Givens epilogue.
  {
    dim3 grid(N_PAIRS / 16, BATCH / 128);
    givens_gemm_kernel<<<grid, 256, 0, stream>>>(
        x, h_prev, wangBf, winBf, b_angle, b_in, decay, out);
  }
}
